// SSM_LM_80547816669368
// MI455X (gfx1250) — matrix operand order repaired and checked by reading, not compiled, not run on hardware
//
#include <hip/hip_runtime.h>
#include <hip/hip_bf16.h>

// ---------------------------------------------------------------------------
// Types for CDNA5 WMMA (gfx1250, wave32): D(16x16 f32) = A(16x32 bf16) x B(32x16 bf16) + C
// ---------------------------------------------------------------------------
typedef __attribute__((ext_vector_type(16))) __bf16 v16bf;
typedef __attribute__((ext_vector_type(8)))  float  v8f;

struct U128 { unsigned x, y, z, w; };

union Frag {
    v16bf          v;
    unsigned       u32v[8];
    unsigned short u16v[16];
    U128           q[2];
};

__device__ inline unsigned short f2bf(float x) {
    unsigned u = __float_as_uint(x);
    unsigned r = u + 0x7FFFu + ((u >> 16) & 1u);   // round-to-nearest-even
    return (unsigned short)(r >> 16);
}

// CDNA5 async global->LDS copy (ASYNCcnt-tracked); lds_off = LDS byte offset.
__device__ inline void async_ld_b128(unsigned lds_off, const void* gaddr) {
    asm volatile("global_load_async_to_lds_b128 %0, %1, off"
                 :: "v"(lds_off), "v"(gaddr) : "memory");
}
__device__ inline void wait_async0() {
    asm volatile("s_wait_asynccnt 0" ::: "memory");
}
__device__ inline void wait_async3() {
    asm volatile("s_wait_asynccnt 3" ::: "memory");
}

// A-fragment (16x32 bf16) from LDS. Layout (contiguous K order, same as B loader):
// lane = row (M = lane&15); lanes<16 hold K={0..15}, lanes>=16 hold K={16..31}.
__device__ inline v16bf load_a_frag_lds(const unsigned short* base, int row_stride, int lane) {
    int m = lane & 15;
    int bk2 = (lane >> 4) * 8;
    const unsigned* r32 = (const unsigned*)(base + m * row_stride);
    Frag f;
#pragma unroll
    for (int v = 0; v < 4; ++v) {
        f.u32v[v]     = r32[bk2 + v];
        f.u32v[4 + v] = r32[4 + bk2 + v];
    }
    return f.v;
}

// B-fragment (32x16 bf16) from LDS. lane = col; 16 contiguous K at K0 = 16*(lane>=16).
__device__ inline v16bf load_b_frag_lds(const unsigned short* base, int row_stride, int lane) {
    int n = lane & 15;
    int k0 = (lane >> 4) * 16;
    const U128* p = (const U128*)(base + n * row_stride + k0);
    Frag f;
    f.q[0] = p[0];
    f.q[1] = p[1];
    return f.v;
}

__device__ inline float warp_sum32(float v) {
#pragma unroll
    for (int m = 1; m < 32; m <<= 1) v += __shfl_xor(v, m, 32);
    return v;
}
__device__ inline float rowmax16(float v) {
#pragma unroll
    for (int m = 1; m < 16; m <<= 1) v = fmaxf(v, __shfl_xor(v, m, 32));
    return v;
}
__device__ inline float rowsum16(float v) {
#pragma unroll
    for (int m = 1; m < 16; m <<= 1) v += __shfl_xor(v, m, 32);
    return v;
}

__device__ inline float block_sum256(float v) {
    __shared__ float sred[8];
#pragma unroll
    for (int m = 1; m < 32; m <<= 1) v += __shfl_xor(v, m, 32);
    int wave = threadIdx.x >> 5, lane = threadIdx.x & 31;
    if (lane == 0) sred[wave] = v;
    __syncthreads();
    float t = (lane < 8) ? sred[lane] : 0.f;
#pragma unroll
    for (int m = 1; m < 8; m <<= 1) t += __shfl_xor(t, m, 32);
    t = __shfl(t, 0, 32);
    __syncthreads();
    return t;
}

__device__ inline float siluf(float x) { return x / (1.f + __expf(-x)); }

// ---------------------------------------------------------------------------
// Model constants
// ---------------------------------------------------------------------------
#define MTOK      4096      // B*T
#define SEQ_T     2048
#define DM        512
#define D_INNER   1024
#define D_STATE   64
#define NH        16
#define D_INPROJ  2192
#define CONVC     1152
#define FFNH      1536
#define AH        8
#define HD        64
#define VOCABN    8192

// ---------------------------------------------------------------------------
// fp32 -> bf16 cast pass (bandwidth bound, once per GEMM operand)
// ---------------------------------------------------------------------------
__global__ __launch_bounds__(256)
void k_cast_bf16(const float* __restrict__ in, unsigned short* __restrict__ out, int n4) {
    int i = blockIdx.x * 256 + threadIdx.x;
    if (i >= n4) return;
    float4 v = ((const float4*)in)[i];
    unsigned lo = (unsigned)f2bf(v.x) | ((unsigned)f2bf(v.y) << 16);
    unsigned hi = (unsigned)f2bf(v.z) | ((unsigned)f2bf(v.w) << 16);
    ((unsigned*)out)[i * 2]     = lo;
    ((unsigned*)out)[i * 2 + 1] = hi;
}

// ---------------------------------------------------------------------------
// WMMA GEMM on bf16 operands:  C[M,N] (+)= scale[n] * (A[M,K] @ W[N,K]^T + bias[n])
// Double-buffered GLOBAL_LOAD_ASYNC_TO_LDS_B128 pipeline; f32 accumulate.
// Block tile 128(M) x 64(N), K-step 32; 8 waves, each wave: 16 rows x 64 cols.
// Every thread issues exactly 3 asyncs per step (W row clamped, not masked),
// so s_wait_asynccnt 3 == "previous tile landed" uniformly per wave.
// ---------------------------------------------------------------------------
#define GTM 128
#define GTN 64
#define GTK 32
#define LDSTR 40   // row pitch in ushorts (80 B, multiple of 16 B)

__global__ __launch_bounds__(256)
void k_gemm_bf(const unsigned short* __restrict__ A, const unsigned short* __restrict__ W,
               float* __restrict__ C, const float* __restrict__ bias,
               const float* __restrict__ scale, int M, int N, int K, int accumulate) {
    __shared__ __align__(16) unsigned short As[2][GTM * LDSTR];
    __shared__ __align__(16) unsigned short Ws[2][GTN * LDSTR];

    const int m0 = blockIdx.y * GTM;
    const int n0 = blockIdx.x * GTN;
    const int tid = threadIdx.x;
    const int lane = tid & 31, wave = tid >> 5;

    // Per-thread staging slots (constant over K):
    // A tile = 128 rows x 4 b128-chunks = 512 chunks -> 2 per thread
    // W tile =  64 rows x 4 b128-chunks = 256 chunks -> 1 per thread
    const int ar0 = tid >> 2,         ac0 = (tid & 3) * 8;
    const int ar1 = (tid + 256) >> 2, ac1 = (tid & 3) * 8;
    const int wr  = tid >> 2,         wc  = (tid & 3) * 8;
    unsigned ldsA0[2], ldsA1[2], ldsW[2];
#pragma unroll
    for (int bsel = 0; bsel < 2; ++bsel) {
        ldsA0[bsel] = (unsigned)(size_t)&As[bsel][ar0 * LDSTR + ac0];
        ldsA1[bsel] = (unsigned)(size_t)&As[bsel][ar1 * LDSTR + ac1];
        ldsW[bsel]  = (unsigned)(size_t)&Ws[bsel][wr * LDSTR + wc];
    }
    int wrow = n0 + wr; if (wrow > N - 1) wrow = N - 1;   // clamp: always issue
    const unsigned short* ga0 = A + (size_t)(m0 + ar0) * K + ac0;
    const unsigned short* ga1 = A + (size_t)(m0 + ar1) * K + ac1;
    const unsigned short* gw  = W + (size_t)wrow * K + wc;

    v8f zero8 = {0.f, 0.f, 0.f, 0.f, 0.f, 0.f, 0.f, 0.f};
    v8f acc[4] = {zero8, zero8, zero8, zero8};

    const int nk = K / GTK;
    // prologue: tile 0 -> buffer 0
    async_ld_b128(ldsA0[0], ga0);
    async_ld_b128(ldsA1[0], ga1);
    async_ld_b128(ldsW[0],  gw);

    for (int k = 0; k < nk; ++k) {
        if (k + 1 < nk) {
            const int nb = (k + 1) & 1;
            const size_t ko = (size_t)(k + 1) * GTK;
            async_ld_b128(ldsA0[nb], ga0 + ko);
            async_ld_b128(ldsA1[nb], ga1 + ko);
            async_ld_b128(ldsW[nb],  gw + ko);
            wait_async3();   // tile k complete (wave's asyncs finish in order)
        } else {
            wait_async0();
        }
        __syncthreads();

        const unsigned short* Ab = As[k & 1];
        const unsigned short* Wb = Ws[k & 1];
        v16bf a = load_a_frag_lds(Ab + wave * 16 * LDSTR, LDSTR, lane);
#pragma unroll
        for (int j = 0; j < 4; ++j) {
            v16bf b = load_b_frag_lds(Wb + j * 16 * LDSTR, LDSTR, lane);
            acc[j] = __builtin_amdgcn_wmma_f32_16x16x32_bf16(
                false, a, false, b, (short)0, acc[j], false, false);
        }
        __syncthreads();   // all reads of buf[k&1] done before it is re-filled
    }

    // Store: C layout — VGPR r -> row r + 8*(lane>=16), col = lane&15
    const int mrow = m0 + wave * 16 + (lane >> 4) * 8;
    const int nc0 = n0 + (lane & 15);
#pragma unroll
    for (int j = 0; j < 4; ++j) {
        int n = nc0 + j * 16;
        if (n >= N) continue;
        float bn = bias ? bias[n] : 0.f;
        float sn = scale ? scale[n] : 1.f;
#pragma unroll
        for (int r = 0; r < 8; ++r) {
            size_t off = (size_t)(mrow + r) * N + n;
            float v = sn * (acc[j][r] + bn);
            if (accumulate) v += C[off];
            C[off] = v;
        }
    }
}

// ---------------------------------------------------------------------------
// Embedding gather
// ---------------------------------------------------------------------------
__global__ __launch_bounds__(256)
void k_gather(const int* __restrict__ ids, const float* __restrict__ tok,
              const float* __restrict__ big, float* __restrict__ embg,
              float* __restrict__ bigg) {
    int t = blockIdx.x;
    int ti = t & (SEQ_T - 1);
    int id = ids[t];
    int prev = ti ? ids[t - 1] : 0;
    int bh = (prev * 1009 + id) % 10240;
    for (int c = threadIdx.x; c < 512; c += 256)
        embg[(size_t)t * 512 + c] = tok[(size_t)id * 512 + c];
    if (threadIdx.x < 128)
        bigg[(size_t)t * 128 + threadIdx.x] = big[(size_t)bh * 128 + threadIdx.x];
}

// ---------------------------------------------------------------------------
// Elementwise kernels
// ---------------------------------------------------------------------------
__global__ __launch_bounds__(256)
void k_mix(float* __restrict__ x, const float* __restrict__ x0,
           const float* __restrict__ rm) {
    int i = blockIdx.x * 256 + threadIdx.x;
    int c = i & (DM - 1);
    x[i] = rm[c] * x[i] + rm[DM + c] * x0[i];
}

__global__ __launch_bounds__(256)
void k_rmsnorm(const float* __restrict__ in, const float* __restrict__ w,
               float* __restrict__ out, int D) {
    size_t row = blockIdx.x;
    const float* r = in + row * D;
    float ss = 0.f;
    for (int c = threadIdx.x; c < D; c += 256) { float v = r[c]; ss += v * v; }
    ss = block_sum256(ss);
    float rr = rsqrtf(ss / (float)D + 1e-6f);
    float* o = out + row * D;
    for (int c = threadIdx.x; c < D; c += 256) o[c] = r[c] * rr * w[c];
}

__global__ __launch_bounds__(256)
void k_conv(const float* __restrict__ zx, const float* __restrict__ cw,
            const float* __restrict__ cb, float* __restrict__ xBC) {
    int i = blockIdx.x * 256 + threadIdx.x;
    int t = i / CONVC, c = i - t * CONVC;
    int ti = t & (SEQ_T - 1);
    float acc = cb[c];
#pragma unroll
    for (int k = 0; k < 4; ++k) {
        int src = ti - 3 + k;
        if (src >= 0)
            acc += zx[(size_t)(t - 3 + k) * D_INPROJ + D_INNER + c] * cw[c * 4 + k];
    }
    xBC[i] = siluf(acc);
}

__global__ __launch_bounds__(256)
void k_silumul(const float* __restrict__ gu, float* __restrict__ h) {
    int i = blockIdx.x * 256 + threadIdx.x;
    int t = i / FFNH, c = i - t * FFNH;
    float g = gu[(size_t)t * (2 * FFNH) + c];
    float u = gu[(size_t)t * (2 * FFNH) + FFNH + c];
    h[i] = siluf(g) * u;
}

__global__ __launch_bounds__(256)
void k_softcap(float* __restrict__ o) {
    int i = blockIdx.x * 256 + threadIdx.x;
    o[i] = 30.0f * tanhf(o[i] * (1.0f / 30.0f));
}

// y = rmsnorm(y * silu(z), norm_w) over D_INNER, in-place on y
__global__ __launch_bounds__(256)
void k_gatenorm(float* __restrict__ y, const float* __restrict__ zx,
                const float* __restrict__ w) {
    size_t row = blockIdx.x;
    float v[4];
    float ss = 0.f;
#pragma unroll
    for (int k = 0; k < 4; ++k) {
        int c = threadIdx.x + k * 256;
        float z = zx[row * D_INPROJ + c];
        float val = y[row * D_INNER + c] * siluf(z);
        v[k] = val; ss += val * val;
    }
    ss = block_sum256(ss);
    float rr = rsqrtf(ss / (float)D_INNER + 1e-6f);
#pragma unroll
    for (int k = 0; k < 4; ++k) {
        int c = threadIdx.x + k * 256;
        y[row * D_INNER + c] = v[k] * rr * w[c];
    }
}

// ---------------------------------------------------------------------------
// Selective scan: one workgroup per (batch, head); 64x64 state in registers.
// ---------------------------------------------------------------------------
__global__ __launch_bounds__(256)
void k_scan(const float* __restrict__ xBC, const float* __restrict__ zx,
            const float* __restrict__ dt_bias, const float* __restrict__ A_log,
            const float* __restrict__ Dp, float* __restrict__ y) {
    __shared__ float sB[64], sC[64], sx[64], sdt[1], sdA[1];
    int bid = blockIdx.x;
    int b = bid >> 4, h = bid & 15;
    int tid = threadIdx.x;
    int p = tid >> 2, jq = tid & 3;

    float Ah = -__expf(A_log[h]);
    float bias = dt_bias[h];
    float Dh = Dp[h];
    float hreg[16];
#pragma unroll
    for (int i = 0; i < 16; ++i) hreg[i] = 0.f;

    for (int ti = 0; ti < SEQ_T; ++ti) {
        size_t row = (size_t)(b * SEQ_T + ti);
        if (tid < 64)       sB[tid]       = xBC[row * CONVC + D_INNER + tid];
        else if (tid < 128) sC[tid - 64]  = xBC[row * CONVC + D_INNER + D_STATE + (tid - 64)];
        else if (tid < 192) sx[tid - 128] = xBC[row * CONVC + h * 64 + (tid - 128)];
        else if (tid == 192) {
            float d0 = zx[row * D_INPROJ + (D_INPROJ - NH) + h] + bias;
            float dt = (d0 > 20.f) ? d0 : log1pf(__expf(d0));
            sdt[0] = dt;
            sdA[0] = __expf(dt * Ah);
        }
        __syncthreads();
        float da = sdA[0];
        float dtx = sdt[0] * sx[p];
        float part = 0.f;
#pragma unroll
        for (int i = 0; i < 16; ++i) {
            hreg[i] = da * hreg[i] + dtx * sB[jq * 16 + i];
            part += hreg[i] * sC[jq * 16 + i];
        }
        part += __shfl_xor(part, 1, 32);
        part += __shfl_xor(part, 2, 32);
        if (jq == 0) y[row * D_INNER + h * 64 + p] = part + Dh * sx[p];
        __syncthreads();
    }
}

// ---------------------------------------------------------------------------
// Attention prep: rmsnorm(q)*gain/8, rmsnorm(k); bf16 out. v stored transposed.
// ---------------------------------------------------------------------------
__global__ __launch_bounds__(256)
void k_qkvprep(const float* __restrict__ qkv, const float* __restrict__ gain,
               unsigned short* __restrict__ qb, unsigned short* __restrict__ kb,
               unsigned short* __restrict__ vT) {
    int row = blockIdx.x * 8 + (threadIdx.x >> 5);
    int lane = threadIdx.x & 31;
    int t = row >> 3, h = row & 7;
    int b = t >> 11, ti = t & (SEQ_T - 1);
    const float* base = qkv + (size_t)t * (3 * DM);

    float q0 = base[h * 64 + lane],          q1 = base[h * 64 + lane + 32];
    float k0 = base[DM + h * 64 + lane],     k1 = base[DM + h * 64 + lane + 32];
    float v0 = base[2 * DM + h * 64 + lane], v1 = base[2 * DM + h * 64 + lane + 32];

    float rq = rsqrtf(warp_sum32(q0 * q0 + q1 * q1) / 64.f + 1e-6f);
    float rk = rsqrtf(warp_sum32(k0 * k0 + k1 * k1) / 64.f + 1e-6f);
    float gs = gain[h] * 0.125f;

    size_t orow = ((size_t)(b * AH + h) * SEQ_T + ti) * 64;
    qb[orow + lane]      = f2bf(q0 * rq * gs);
    qb[orow + lane + 32] = f2bf(q1 * rq * gs);
    kb[orow + lane]      = f2bf(k0 * rk);
    kb[orow + lane + 32] = f2bf(k1 * rk);
    size_t vb = ((size_t)(b * AH + h) * 64) * SEQ_T + ti;
    vT[vb + (size_t)lane * SEQ_T]        = f2bf(v0);
    vT[vb + (size_t)(lane + 32) * SEQ_T] = f2bf(v1);
}

// ---------------------------------------------------------------------------
// Flash attention: one wave per (bh, 16-query tile). QK^T and P@V on WMMA.
// ---------------------------------------------------------------------------
__global__ __launch_bounds__(32)
void k_attn(const unsigned short* __restrict__ qb, const unsigned short* __restrict__ kb,
            const unsigned short* __restrict__ vT, float* __restrict__ o) {
    __shared__ __align__(16) unsigned short Pl[16 * LDSTR];
    const int lane = threadIdx.x;
    const int q0 = blockIdx.x * 16;
    const int bh = blockIdx.y;
    const int b = bh >> 3, h = bh & 7;
    const int mrow = lane & 15, hi = lane >> 4;

    const unsigned short* qrow0 = qb + ((size_t)bh * SEQ_T) * 64;
    const unsigned short* krow0 = kb + ((size_t)bh * SEQ_T) * 64;
    const unsigned short* vrow0 = vT + ((size_t)bh * 64) * SEQ_T;

    Frag aq0, aq1;
    {
        const unsigned* r32 = (const unsigned*)(qrow0 + (size_t)(q0 + mrow) * 64);
        int bk2 = hi * 8;
#pragma unroll
        for (int v = 0; v < 4; ++v) {
            aq0.u32v[v]     = r32[bk2 + v];
            aq0.u32v[4 + v] = r32[4 + bk2 + v];
            aq1.u32v[v]     = r32[16 + bk2 + v];
            aq1.u32v[4 + v] = r32[20 + bk2 + v];
        }
    }

    v8f zero8 = {0.f, 0.f, 0.f, 0.f, 0.f, 0.f, 0.f, 0.f};
    v8f oacc[4] = {zero8, zero8, zero8, zero8};
    float mrun[8], lrun[8];
#pragma unroll
    for (int r = 0; r < 8; ++r) { mrun[r] = -3.0e38f; lrun[r] = 0.f; }

    const int nkb = (q0 + 15) / 32 + 1;
    for (int kbv = 0; kbv < nkb; ++kbv) {
        const int s0 = kbv * 32;
        v8f sc[2];
#pragma unroll
        for (int sub = 0; sub < 2; ++sub) {
            const unsigned short* kr = krow0 + (size_t)(s0 + sub * 16 + mrow) * 64 + hi * 16;
            Frag bk0, bk1;
            const U128* p0 = (const U128*)kr;
            const U128* p1 = (const U128*)(kr + 32);
            bk0.q[0] = p0[0]; bk0.q[1] = p0[1];
            bk1.q[0] = p1[0]; bk1.q[1] = p1[1];
            v8f z = zero8;
            z = __builtin_amdgcn_wmma_f32_16x16x32_bf16(false, aq0.v, false, bk0.v, (short)0, z, false, false);
            z = __builtin_amdgcn_wmma_f32_16x16x32_bf16(false, aq1.v, false, bk1.v, (short)0, z, false, false);
            sc[sub] = z;
        }
#pragma unroll
        for (int r = 0; r < 8; ++r) {
            int trow = q0 + r + 8 * hi;
            bool m0 = (s0 + mrow) > trow;
            bool m1 = (s0 + 16 + mrow) > trow;
            float v0 = sc[0][r], v1 = sc[1][r];
            float mx = fmaxf(m0 ? -3.0e38f : v0, m1 ? -3.0e38f : v1);
            mx = rowmax16(mx);
            float mnew = fmaxf(mrun[r], mx);
            float p0 = m0 ? 0.f : __expf(v0 - mnew);
            float p1 = m1 ? 0.f : __expf(v1 - mnew);
            float corr = __expf(mrun[r] - mnew);
            float rs = rowsum16(p0 + p1);
            lrun[r] = lrun[r] * corr + rs;
            mrun[r] = mnew;
#pragma unroll
            for (int j = 0; j < 4; ++j) oacc[j][r] *= corr;
            Pl[(r + 8 * hi) * LDSTR + mrow]      = f2bf(p0);
            Pl[(r + 8 * hi) * LDSTR + 16 + mrow] = f2bf(p1);
        }
        __syncthreads();
        v16bf pa = load_a_frag_lds(Pl, LDSTR, lane);
#pragma unroll
        for (int j = 0; j < 4; ++j) {
            const unsigned short* vr = vrow0 + (size_t)(j * 16 + mrow) * SEQ_T + s0 + hi * 16;
            Frag bv;
            const U128* p = (const U128*)vr;
            bv.q[0] = p[0]; bv.q[1] = p[1];
            oacc[j] = __builtin_amdgcn_wmma_f32_16x16x32_bf16(false, pa, false, bv.v, (short)0, oacc[j], false, false);
        }
        __syncthreads();
    }
#pragma unroll
    for (int j = 0; j < 4; ++j) {
#pragma unroll
        for (int r = 0; r < 8; ++r) {
            int trow = q0 + r + 8 * hi;
            o[(size_t)(b * SEQ_T + trow) * DM + h * 64 + j * 16 + mrow] = oacc[j][r] / lrun[r];
        }
    }
}

// ---------------------------------------------------------------------------
// Host orchestration
// ---------------------------------------------------------------------------
extern "C" void kernel_launch(void* const* d_in, const int* in_sizes, int n_in,
                              void* d_out, int out_size, void* d_ws, size_t ws_size,
                              hipStream_t stream) {
    (void)in_sizes; (void)n_in; (void)out_size; (void)ws_size;
    const int* ids = (const int*)d_in[0];
    auto F = [&](int i) { return (const float*)d_in[i]; };

    char* ws = (char*)d_ws;
    size_t off = 0;
    auto alloc = [&](size_t bytes) -> void* {
        void* p = ws + off;
        off = (off + bytes + 255) & ~(size_t)255;
        return p;
    };
    float* x    = (float*)alloc((size_t)MTOK * DM * 4);
    float* x0   = (float*)alloc((size_t)MTOK * DM * 4);
    float* xn   = (float*)alloc((size_t)MTOK * DM * 4);
    float* big0 = (float*)alloc((size_t)MTOK * 3072 * 4);    // zxbcdt / gu / qkv
    float* big1 = (float*)alloc((size_t)MTOK * FFNH * 4);    // xBC / ffn hidden
    float* ybuf = (float*)alloc((size_t)MTOK * D_INNER * 4); // scan y / attn o / e
    unsigned short* abf = (unsigned short*)alloc((size_t)MTOK * FFNH * 2);   // bf16 A stage
    unsigned short* wbf = (unsigned short*)alloc((size_t)VOCABN * DM * 2);   // bf16 W stage
    // bf16 q/k/vT live in unused tail of big0 (qkv only uses first 25.2 MB)
    unsigned short* qb = (unsigned short*)((char*)big0 + 33554432);
    unsigned short* kb = qb + (size_t)MTOK * DM;
    unsigned short* vT = kb + (size_t)MTOK * DM;
    float* embg = big0;   // only used before the layer stack
    float* bigg = big1;

    auto cast = [&](const float* src, unsigned short* dst, size_t n) {
        int n4 = (int)(n >> 2);
        k_cast_bf16<<<(n4 + 255) / 256, 256, 0, stream>>>(src, dst, n4);
    };
    auto gemm = [&](const float* Af, const float* Wf, float* Cm, const float* bias,
                    const float* scale, int N, int K, bool acc) {
        cast(Af, abf, (size_t)MTOK * K);
        cast(Wf, wbf, (size_t)N * K);
        dim3 g((N + GTN - 1) / GTN, MTOK / GTM);
        k_gemm_bf<<<g, 256, 0, stream>>>(abf, wbf, Cm, bias, scale, MTOK, N, K, acc ? 1 : 0);
    };
    auto rms = [&](const float* in, const float* w, float* out, int D) {
        k_rmsnorm<<<MTOK, 256, 0, stream>>>(in, w, out, D);
    };

    // ---- embedding ----
    k_gather<<<MTOK, 256, 0, stream>>>(ids, F(1), F(6), embg, bigg);
    gemm(embg, F(2), x, nullptr, nullptr, DM, 512, false);
    gemm(bigg, F(7), x, nullptr, nullptr, DM, 128, true);
    hipMemcpyAsync(x0, x, (size_t)MTOK * DM * 4, hipMemcpyDeviceToDevice, stream);

    int js = 0;
    for (int L = 0; L < 10; ++L) {
        if (L == 6) {
            // ---- attention block ----
            const int ab = 144;
            k_mix<<<MTOK * DM / 256, 256, 0, stream>>>(x, x0, F(ab + 0));
            rms(x, F(ab + 1), xn, DM);
            gemm(xn, F(ab + 2), big0, nullptr, nullptr, 3 * DM, DM, false);  // qkv
            k_qkvprep<<<MTOK * AH / 8, 256, 0, stream>>>(big0, F(ab + 4), qb, kb, vT);
            k_attn<<<dim3(SEQ_T / 16, 2 * AH), 32, 0, stream>>>(qb, kb, vT, ybuf);
            gemm(ybuf, F(ab + 3), x, nullptr, F(ab + 5), DM, DM, true);
            rms(x, F(ab + 6), xn, DM);
            gemm(xn, F(ab + 8), big0, nullptr, nullptr, 2 * FFNH, DM, false);
            k_silumul<<<MTOK * FFNH / 256, 256, 0, stream>>>(big0, big1);
            gemm(big1, F(ab + 9), x, nullptr, F(ab + 7), DM, FFNH, true);
        } else {
            // ---- SSM block ----
            const int sb = 9 + 15 * js; ++js;
            k_mix<<<MTOK * DM / 256, 256, 0, stream>>>(x, x0, F(sb + 0));
            rms(x, F(sb + 1), xn, DM);
            gemm(xn, F(sb + 3), big0, nullptr, nullptr, D_INPROJ, DM, false);   // zxbcdt
            k_conv<<<MTOK * CONVC / 256, 256, 0, stream>>>(big0, F(sb + 4), F(sb + 5), big1);
            k_scan<<<2 * NH, 256, 0, stream>>>(big1, big0, F(sb + 6), F(sb + 7), F(sb + 8), ybuf);
            k_gatenorm<<<MTOK, 256, 0, stream>>>(ybuf, big0, F(sb + 9));
            gemm(ybuf, F(sb + 10), x, nullptr, F(sb + 2), DM, D_INNER, true);
            rms(x, F(sb + 11), xn, DM);
            gemm(xn, F(sb + 13), big0, nullptr, nullptr, 2 * FFNH, DM, false);
            k_silumul<<<MTOK * FFNH / 256, 256, 0, stream>>>(big0, big1);
            gemm(big1, F(sb + 14), x, nullptr, F(sb + 12), DM, FFNH, true);
        }
    }

    // ---- head ----
    rms(x, F(8), xn, DM);
    gemm(xn, F(3), ybuf, nullptr, nullptr, 512, DM, false);
    gemm(ybuf, F(4), (float*)d_out, F(5), nullptr, VOCABN, 512, false);
    k_softcap<<<(size_t)MTOK * VOCABN / 256, 256, 0, stream>>>((float*)d_out);
}
